// TTransformer_69277822484805
// MI455X (gfx1250) — matrix operand order repaired and checked by reading, not compiled, not run on hardware
//
#include <hip/hip_runtime.h>
#include <hip/hip_bf16.h>
#include <math.h>

// ---------------- problem constants ----------------
#define BB 16
#define NN 307
#define TT 24
#define CC 128
#define HH 8
#define DD 16
#define SS 20   // S = min(5*ceil(log(24)), 24) = 20
#define UU 20
#define ROWS (BB * NN * TT)          // 117888, multiple of 16
#define ROWBLK (ROWS / 16)           // 7368
#define FF 512                       // 4*C
#define NT (NN * TT)                 // 7368

typedef __attribute__((ext_vector_type(16))) __bf16 v16bf;
typedef __attribute__((ext_vector_type(8)))  __bf16 v8bf;
typedef __attribute__((ext_vector_type(8)))  float  v8f;
typedef __attribute__((ext_vector_type(4)))  unsigned int v4u;
typedef __attribute__((ext_vector_type(8)))  int v8i;
typedef __attribute__((ext_vector_type(4)))  int v4i;

static __device__ __forceinline__ v8f wmma_bf16(v16bf a, v16bf b, v8f c) {
  return __builtin_amdgcn_wmma_f32_16x16x32_bf16(false, a, false, b, (short)0, c, false, false);
}

// ---------------- TDM: 1-D tensor_load_to_lds (n8 units of 8 bytes) ----------
// D# per CDNA5 ISA: group0 = {count=1 | lds_addr | global_addr | type=2},
// group1 = {data_size=8B, tensor_dim0=tile_dim0=n8, stride=n8, 1-D}.
static __device__ __forceinline__ void tdm_load_1d(unsigned lds_off,
                                                   const void* gsrc,
                                                   unsigned n8) {
  const unsigned long long ga = (unsigned long long)gsrc;
  v4u g0;
  g0.x = 1u;                                              // count=1 (valid)
  g0.y = lds_off;                                         // lds_addr
  g0.z = (unsigned)ga;                                    // global_addr[31:0]
  g0.w = (unsigned)((ga >> 32) & 0x01FFFFFFull) | (2u << 30); // addr[56:32]|type=2
  v8i g1;
  g1[0] = (int)(3u << 16);                                // data_size = 8 bytes
  g1[1] = (int)((n8 & 0xFFFFu) << 16);                    // tensor_dim0 lo16
  g1[2] = (int)(((n8 >> 16) & 0xFFFFu) | (1u << 16));     // dim0 hi16 | tensor_dim1=1
  g1[3] = (int)((n8 & 0xFFFFu) << 16);                    // tile_dim0 = n8
  g1[4] = 0;                                              // tile_dim1/2 unused
  g1[5] = (int)n8;                                        // tensor_dim0_stride
  g1[6] = 0;
  g1[7] = 0;
  v4i gz = {0, 0, 0, 0};
#if defined(__clang_major__) && (__clang_major__ >= 23)
  v8i gz8 = {0, 0, 0, 0, 0, 0, 0, 0};
  __builtin_amdgcn_tensor_load_to_lds(g0, g1, gz, gz, gz8, 0);
#else
  __builtin_amdgcn_tensor_load_to_lds(g0, g1, gz, gz, 0);
#endif
}

// A-matrix (16x32 bf16) lane layout: lane holds row m=lane&15,
// K chunks [o, o+8) and [o+16, o+24), o = kt*32 + (lane>>4)*8.
static __device__ __forceinline__ v16bf load_a_bf16(const __bf16* __restrict__ row_base,
                                                    int ldr, int kt, int lane) {
  const int m = lane & 15;
  const int o = kt * 32 + ((lane >> 4) << 3);
  const __bf16* p = row_base + m * ldr + o;
  union { v16bf v; v8bf h[2]; } u;
  u.h[0] = *(const v8bf*)(p);
  u.h[1] = *(const v8bf*)(p + 16);
  return u.v;
}

// ---------------- kernel 1: te + positional embedding (B*T rows of C) --------
__global__ __launch_bounds__(128) void tepe_kernel(
    const int* __restrict__ time_enc, const float* __restrict__ emb_min,
    const float* __restrict__ emb_hr, const float* __restrict__ emb_wd,
    const float* __restrict__ emb_mo, const float* __restrict__ emb_yr,
    float* __restrict__ tepe) {
  const int bt = blockIdx.x;            // b*T + t
  const int t  = bt % TT;
  const int c  = threadIdx.x;
  const int* te = time_enc + bt * 5;
  float v = emb_min[te[0] * CC + c] + emb_hr[te[1] * CC + c] +
            emb_wd[te[2] * CC + c] + emb_mo[te[3] * CC + c] +
            emb_yr[te[4] * CC + c];
  const int i = c >> 1;
  const float div = __expf((float)(2 * i) * (-logf(10000.0f) / (float)CC));
  const float ang = (float)t * div;
  v += (c & 1) ? __cosf(ang) : __sinf(ang);
  tepe[bt * CC + c] = v;
}

// ---------------- kernel 2: pack (K x Ncols) f32 weight into WMMA B order ----
// out[((ntile*KT + kt)*32 + lane)*16 + j], K(lane,j) = kt*32 + (lane>>4)*8 + (j&7) + (j>>3)*16
__global__ __launch_bounds__(256) void pack_w_kernel(
    const float* __restrict__ W, __bf16* __restrict__ out, int K, int Ncols) {
  const int idx = blockIdx.x * blockDim.x + threadIdx.x;
  if (idx >= K * Ncols) return;
  const int j    = idx & 15;
  const int lane = (idx >> 4) & 31;
  const int rest = idx >> 9;
  const int KT   = K >> 5;
  const int kt   = rest % KT;
  const int nt   = rest / KT;
  const int n    = nt * 16 + (lane & 15);
  const int k    = kt * 32 + ((lane >> 4) << 3) + (j & 7) + ((j >> 3) << 4);
  out[idx] = (__bf16)W[k * Ncols + n];
}

// ---------------- kernel 3: fused x-build + QKV projection -------------------
// A tile (16x128) = query + tepe, converted once per block into LDS in
// WMMA A-lane-packed order; each wave then feeds WMMA from ds_load_b128.
// All row->(b,n,t) decompositions use block-uniform bases (SALU) + wrap fixups.
__global__ __launch_bounds__(256) void qkv_kernel(
    const float* __restrict__ query, const float* __restrict__ tepe,
    const __bf16* __restrict__ wq, const __bf16* __restrict__ wk,
    const __bf16* __restrict__ wv,
    __bf16* __restrict__ qo, __bf16* __restrict__ ko, __bf16* __restrict__ vo) {
  __shared__ __attribute__((aligned(32))) __bf16 aPack[4 * 32 * 16];  // 4 KB
  const int tid = threadIdx.x;
  const int wave = tid >> 5;
  const int lane = tid & 31;
  const int rowbase = blockIdx.x * 16;
  // block-uniform decomposition of rowbase (SALU divisions, once per block)
  const int t_base = rowbase % TT;
  const int nb     = rowbase / TT;
  const int n_base = nb % NN;
  const int b_base = nb / NN;
  const int rm     = rowbase % NT;

  { // cooperative stage: x = query + (pe+te), packed A layout, 8 elems/thread
    const int kt   = tid >> 6;
    const int ln   = (tid >> 1) & 31;
    const int half = tid & 1;
    const int m    = ln & 15;
    const int ko2  = kt * 32 + ((ln >> 4) << 3) + half * 16;
    int t = t_base + m;
    if (t >= TT) t -= TT;
    const int b = b_base + ((rm + m) >= NT ? 1 : 0);
    const float* qp = query + (rowbase + m) * CC + ko2;
    const float* tp = tepe + (b * TT + t) * CC + ko2;
    __bf16* dst = &aPack[(kt * 32 + ln) * 16 + half * 8];
#pragma unroll
    for (int j = 0; j < 8; ++j) dst[j] = (__bf16)(qp[j] + tp[j]);
  }
  __syncthreads();

  v16bf a[4];
#pragma unroll
  for (int kt = 0; kt < 4; ++kt) a[kt] = *(const v16bf*)&aPack[(kt * 32 + lane) * 16];

  v8f aq = {}, ak = {}, av = {};
#pragma unroll
  for (int kt = 0; kt < 4; ++kt) {
    const int bidx = ((wave * 4 + kt) * 32 + lane) * 16;
    aq = wmma_bf16(a[kt], *(const v16bf*)(wq + bidx), aq);
    ak = wmma_bf16(a[kt], *(const v16bf*)(wk + bidx), ak);
    av = wmma_bf16(a[kt], *(const v16bf*)(wv + bidx), av);
  }

  const int m_off = (lane >> 4) << 3;
  const int nl = lane & 15;            // == d
  const int h = wave;                  // col tile == head
#pragma unroll
  for (int i = 0; i < 8; ++i) {
    const int rr = i + m_off;          // 0..15 within block
    int t = t_base + rr;
    int n = n_base, b = b_base;
    if (t >= TT) { t -= TT; ++n; if (n >= NN) { n = 0; ++b; } }
    const int o = (((b * HH + h) * NN + n) * TT + t) * DD + nl;
    qo[o] = (__bf16)aq[i];
    ko[o] = (__bf16)ak[i];
    vo[o] = (__bf16)av[i];
  }
}

// ---------------- kernel 4: ProbSparse attention per (b,h,n) -----------------
__global__ __launch_bounds__(128) void attn_kernel(
    const __bf16* __restrict__ q, const __bf16* __restrict__ k,
    const __bf16* __restrict__ v, const int* __restrict__ index_sample,
    __bf16* __restrict__ ctx) {
  __shared__ float Qs[TT][DD], Ks[TT][DD], Vs[TT][DD];
  __shared__ int   idx[TT * SS];
  __shared__ float qk[TT * SS], Ms[TT], sc[UU * TT], upd[UU][DD], Vmean[DD];
  __shared__ int   Mtop[UU], pos[TT];

  const int tid = threadIdx.x;
  const int base = blockIdx.x * TT * DD;

  for (int p = tid; p < TT * DD; p += 128) {
    Qs[p / DD][p % DD] = (float)q[base + p];
    Ks[p / DD][p % DD] = (float)k[base + p];
    Vs[p / DD][p % DD] = (float)v[base + p];
  }
  for (int p = tid; p < TT * SS; p += 128) idx[p] = index_sample[p];
  __syncthreads();

  for (int p = tid; p < TT * SS; p += 128) {
    const int t = p / SS;
    const int ks = idx[p];
    float acc = 0.f;
#pragma unroll
    for (int d = 0; d < DD; ++d) acc += Qs[t][d] * Ks[ks][d];
    qk[p] = acc;
  }
  __syncthreads();

  if (tid < TT) {
    float mx = -1e30f, sm = 0.f;
#pragma unroll
    for (int s = 0; s < SS; ++s) {
      const float val = qk[tid * SS + s];
      mx = fmaxf(mx, val); sm += val;
    }
    Ms[tid] = mx - sm * (1.0f / SS);
    pos[tid] = -1;
  }
  __syncthreads();

  if (tid == 0) {                       // top-U selection over 24 entries
    unsigned used = 0;
    for (int u = 0; u < UU; ++u) {
      float best = -1e30f; int bi = 0;
      for (int t2 = 0; t2 < TT; ++t2)
        if (!((used >> t2) & 1u) && Ms[t2] > best) { best = Ms[t2]; bi = t2; }
      used |= 1u << bi;
      Mtop[u] = bi;
      pos[bi] = u;
    }
  }
  __syncthreads();

  for (int p = tid; p < UU * TT; p += 128) {
    const int u = p / TT, t = p % TT;
    const int qi = Mtop[u];
    float acc = 0.f;
#pragma unroll
    for (int d = 0; d < DD; ++d) acc += Qs[qi][d] * Ks[t][d];
    sc[p] = acc * 0.25f;               // 1/sqrt(D)
  }
  __syncthreads();

  if (tid < UU) {
    float mx = -1e30f;
#pragma unroll
    for (int t = 0; t < TT; ++t) mx = fmaxf(mx, sc[tid * TT + t]);
    float den = 0.f;
#pragma unroll
    for (int t = 0; t < TT; ++t) {
      const float e = __expf(sc[tid * TT + t] - mx);
      sc[tid * TT + t] = e; den += e;
    }
    const float inv = 1.0f / den;
#pragma unroll
    for (int d = 0; d < DD; ++d) {
      float acc = 0.f;
#pragma unroll
      for (int t = 0; t < TT; ++t) acc += sc[tid * TT + t] * Vs[t][d];
      upd[tid][d] = acc * inv;
    }
  }
  if (tid < DD) {
    float acc = 0.f;
#pragma unroll
    for (int t = 0; t < TT; ++t) acc += Vs[t][tid];
    Vmean[tid] = acc * (1.0f / TT);
  }
  __syncthreads();

  const int n  = blockIdx.x % NN;
  const int bh = blockIdx.x / NN;
  const int h  = bh % HH, b = bh / HH;
  const int obase = ((b * NN + n) * TT) * CC + h * DD;
  for (int p = tid; p < TT * DD; p += 128) {
    const int t = p / DD, d = p % DD;
    const float val = (pos[t] >= 0) ? upd[pos[t]][d] : Vmean[d];
    ctx[obase + t * CC + d] = (__bf16)val;
  }
}

// ---------------- kernel 5: att_out = ctx@Wo + bo; x1 = LN1(att_out + x) -----
// ctx tile staged via TDM (tensor_load_to_lds); residual x recomputed as
// query + tepe (x is never materialized in global memory).
__global__ __launch_bounds__(256) void oproj_ln1_kernel(
    const __bf16* __restrict__ ctx, const __bf16* __restrict__ wo,
    const float* __restrict__ bo, const float* __restrict__ query,
    const float* __restrict__ tepe, const float* __restrict__ g,
    const float* __restrict__ bta, float* __restrict__ x1) {
  __shared__ __attribute__((aligned(32))) __bf16 ctxs[16 * CC];   // 4 KB
  __shared__ float ys[16][CC];
  const int wave = threadIdx.x >> 5;
  const int lane = threadIdx.x & 31;
  const int rowbase = blockIdx.x * 16;
  const int t_base = rowbase % TT;       // block-uniform (SALU)
  const int b_base = rowbase / NT;
  const int rm     = rowbase % NT;

  if (threadIdx.x < 32) {   // wave 0 issues the tensor DMA (EXEC-independent)
    tdm_load_1d((unsigned)(unsigned long long)&ctxs[0],
                ctx + rowbase * CC, (16 * CC * 2) / 8);
    __builtin_amdgcn_s_wait_tensorcnt(0);
  }
  __syncthreads();

  v8f acc = {};
#pragma unroll
  for (int kt = 0; kt < 4; ++kt) {
    v16bf a = load_a_bf16(ctxs, CC, kt, lane);
    const int bidx = ((wave * 4 + kt) * 32 + lane) * 16;
    acc = wmma_bf16(a, *(const v16bf*)(wo + bidx), acc);
  }

  const int m_off = (lane >> 4) << 3;
  const int nl = lane & 15;
  const float* qrow = query + rowbase * CC;
#pragma unroll
  for (int i = 0; i < 8; ++i) {
    const int rr = i + m_off;
    const int c = wave * 16 + nl;
    int t = t_base + rr;
    if (t >= TT) t -= TT;
    const int b = b_base + ((rm + rr) >= NT ? 1 : 0);
    const float xv = qrow[rr * CC + c] + tepe[(b * TT + t) * CC + c];
    ys[rr][c] = acc[i] + bo[c] + xv;
  }
  __syncthreads();

  // LayerNorm: one row per 16-lane half-wave
  const int row = wave * 2 + (lane >> 4);
  const int seg = (lane & 15) * 8;
  float s = 0.f, s2 = 0.f;
#pragma unroll
  for (int j = 0; j < 8; ++j) {
    const float vv = ys[row][seg + j];
    s += vv; s2 += vv * vv;
  }
#pragma unroll
  for (int m = 1; m < 16; m <<= 1) {
    s  += __shfl_xor(s,  m, 16);
    s2 += __shfl_xor(s2, m, 16);
  }
  const float mean = s * (1.0f / CC);
  const float var  = s2 * (1.0f / CC) - mean * mean;
  const float rstd = rsqrtf(var + 1e-5f);
  const int rbase = (rowbase + row) * CC;
#pragma unroll
  for (int j = 0; j < 8; ++j) {
    const int c = seg + j;
    x1[rbase + c] = (ys[row][c] - mean) * rstd * g[c] + bta[c];
  }
}

// ---------------- kernel 6: out = LN2(relu(x1@W1+b1)@W2 + b2 + x1) -----------
// x1 tile staged via TDM into LDS (f32), converted once into packed-A bf16.
__global__ __launch_bounds__(256) void ffn_ln2_kernel(
    const float* __restrict__ x1, const __bf16* __restrict__ w1p,
    const float* __restrict__ b1, const __bf16* __restrict__ w2p,
    const float* __restrict__ b2, const float* __restrict__ g,
    const float* __restrict__ bta, float* __restrict__ out) {
  __shared__ __attribute__((aligned(32))) float  xs[16 * CC];      // 8 KB
  __shared__ __attribute__((aligned(32))) __bf16 aPack[4 * 32 * 16]; // 4 KB
  __shared__ __attribute__((aligned(32))) __bf16 hid[16 * FF];     // 16 KB
  __shared__ float ys[16][CC];
  const int tid = threadIdx.x;
  const int wave = tid >> 5;
  const int lane = tid & 31;
  const int rowbase = blockIdx.x * 16;

  if (tid < 32) {
    tdm_load_1d((unsigned)(unsigned long long)&xs[0],
                x1 + rowbase * CC, (16 * CC * 4) / 8);
    __builtin_amdgcn_s_wait_tensorcnt(0);
  }
  __syncthreads();

  { // convert staged f32 tile into packed-A bf16 (once per block)
    const int kt   = tid >> 6;
    const int ln   = (tid >> 1) & 31;
    const int half = tid & 1;
    const int m    = ln & 15;
    const int ko2  = kt * 32 + ((ln >> 4) << 3) + half * 16;
    const float* sp = &xs[m * CC + ko2];
    __bf16* dst = &aPack[(kt * 32 + ln) * 16 + half * 8];
#pragma unroll
    for (int j = 0; j < 8; ++j) dst[j] = (__bf16)sp[j];
  }
  __syncthreads();

  v16bf a[4];
#pragma unroll
  for (int kt = 0; kt < 4; ++kt) a[kt] = *(const v16bf*)&aPack[(kt * 32 + lane) * 16];

  const int m_off = (lane >> 4) << 3;
  const int nl = lane & 15;

  // GEMM1: 128 -> 512, 4 col tiles per wave, ReLU into LDS (bf16)
#pragma unroll
  for (int tt = 0; tt < 4; ++tt) {
    const int nt = wave * 4 + tt;           // 0..31
    v8f acc = {};
#pragma unroll
    for (int kt = 0; kt < 4; ++kt) {
      const int bidx = ((nt * 4 + kt) * 32 + lane) * 16;
      acc = wmma_bf16(a[kt], *(const v16bf*)(w1p + bidx), acc);
    }
#pragma unroll
    for (int i = 0; i < 8; ++i) {
      const int mrow = i + m_off;
      const int c = nt * 16 + nl;
      hid[mrow * FF + c] = (__bf16)fmaxf(acc[i] + b1[c], 0.0f);
    }
  }
  __syncthreads();

  // GEMM2: 512 -> 128, one col tile per wave, K = 512 (16 k-tiles from LDS)
  v8f acc2 = {};
#pragma unroll
  for (int kt = 0; kt < 16; ++kt) {
    v16bf a2 = load_a_bf16(hid, FF, kt, lane);
    const int bidx = ((wave * 16 + kt) * 32 + lane) * 16;
    acc2 = wmma_bf16(a2, *(const v16bf*)(w2p + bidx), acc2);
  }
#pragma unroll
  for (int i = 0; i < 8; ++i) {
    const int mrow = i + m_off;
    const int c = wave * 16 + nl;
    ys[mrow][c] = acc2[i] + b2[c] + xs[mrow * CC + c];
  }
  __syncthreads();

  // LayerNorm 2 -> final output
  const int row = wave * 2 + (lane >> 4);
  const int seg = (lane & 15) * 8;
  float s = 0.f, s2 = 0.f;
#pragma unroll
  for (int j = 0; j < 8; ++j) {
    const float vv = ys[row][seg + j];
    s += vv; s2 += vv * vv;
  }
#pragma unroll
  for (int m = 1; m < 16; m <<= 1) {
    s  += __shfl_xor(s,  m, 16);
    s2 += __shfl_xor(s2, m, 16);
  }
  const float mean = s * (1.0f / CC);
  const float var  = s2 * (1.0f / CC) - mean * mean;
  const float rstd = rsqrtf(var + 1e-5f);
  const int rbase = (rowbase + row) * CC;
#pragma unroll
  for (int j = 0; j < 8; ++j) {
    const int c = seg + j;
    out[rbase + c] = (ys[row][c] - mean) * rstd * g[c] + bta[c];
  }
}

// ---------------- launcher ----------------
extern "C" void kernel_launch(void* const* d_in, const int* in_sizes, int n_in,
                              void* d_out, int out_size, void* d_ws, size_t ws_size,
                              hipStream_t stream) {
  const float* query     = (const float*)d_in[2];
  const int*   time_enc  = (const int*)d_in[4];
  const int*   index_smp = (const int*)d_in[5];
  const float* emb_min   = (const float*)d_in[6];
  const float* emb_hr    = (const float*)d_in[7];
  const float* emb_wd    = (const float*)d_in[8];
  const float* emb_mo    = (const float*)d_in[9];
  const float* emb_yr    = (const float*)d_in[10];
  const float* Wq        = (const float*)d_in[11];
  const float* Wk        = (const float*)d_in[12];
  const float* Wv        = (const float*)d_in[13];
  const float* Wo        = (const float*)d_in[14];
  const float* bo        = (const float*)d_in[15];
  const float* ff_w1     = (const float*)d_in[16];
  const float* ff_b1     = (const float*)d_in[17];
  const float* ff_w2     = (const float*)d_in[18];
  const float* ff_b2     = (const float*)d_in[19];
  const float* ln1_g     = (const float*)d_in[20];
  const float* ln1_b     = (const float*)d_in[21];
  const float* ln2_g     = (const float*)d_in[22];
  const float* ln2_b     = (const float*)d_in[23];
  float* out = (float*)d_out;

  char* wsb = (char*)d_ws;
  // workspace layout (bytes, 256-aligned); x never materialized.
  const size_t off_tepe = 0;                           // 196,608
  const size_t off_wq   = 196608;
  const size_t off_wk   = off_wq + 32768;
  const size_t off_wv   = off_wk + 32768;
  const size_t off_wo   = off_wv + 32768;
  const size_t off_w1   = off_wo + 32768;              // 131,072
  const size_t off_w2   = off_w1 + 131072;             // 131,072
  const size_t off_q    = off_w2 + 131072;             // bf16 Q/K/V (30.2 MB ea)
  const size_t off_k    = off_q  + (size_t)ROWS * CC * 2;
  const size_t off_v    = off_k  + (size_t)ROWS * CC * 2;
  const size_t off_ctx  = off_v  + (size_t)ROWS * CC * 2;
  const size_t off_x1   = off_k;   // aliases K+V (dead after attention), exact fit

  float*  tepe = (float*)(wsb + off_tepe);
  __bf16* wqp  = (__bf16*)(wsb + off_wq);
  __bf16* wkp  = (__bf16*)(wsb + off_wk);
  __bf16* wvp  = (__bf16*)(wsb + off_wv);
  __bf16* wop  = (__bf16*)(wsb + off_wo);
  __bf16* w1p  = (__bf16*)(wsb + off_w1);
  __bf16* w2p  = (__bf16*)(wsb + off_w2);
  __bf16* qb   = (__bf16*)(wsb + off_q);
  __bf16* kb   = (__bf16*)(wsb + off_k);
  __bf16* vb   = (__bf16*)(wsb + off_v);
  __bf16* ctx  = (__bf16*)(wsb + off_ctx);
  float*  x1   = (float*)(wsb + off_x1);

  tepe_kernel<<<BB * TT, 128, 0, stream>>>(time_enc, emb_min, emb_hr, emb_wd,
                                           emb_mo, emb_yr, tepe);
  pack_w_kernel<<<(CC * CC + 255) / 256, 256, 0, stream>>>(Wq, wqp, CC, CC);
  pack_w_kernel<<<(CC * CC + 255) / 256, 256, 0, stream>>>(Wk, wkp, CC, CC);
  pack_w_kernel<<<(CC * CC + 255) / 256, 256, 0, stream>>>(Wv, wvp, CC, CC);
  pack_w_kernel<<<(CC * CC + 255) / 256, 256, 0, stream>>>(Wo, wop, CC, CC);
  pack_w_kernel<<<(CC * FF + 255) / 256, 256, 0, stream>>>(ff_w1, w1p, CC, FF);
  pack_w_kernel<<<(FF * CC + 255) / 256, 256, 0, stream>>>(ff_w2, w2p, FF, CC);

  qkv_kernel<<<ROWBLK, 256, 0, stream>>>(query, tepe, wqp, wkp, wvp, qb, kb, vb);

  attn_kernel<<<BB * HH * NN, 128, 0, stream>>>(qb, kb, vb, index_smp, ctx);

  oproj_ln1_kernel<<<ROWBLK, 256, 0, stream>>>(ctx, wop, bo, query, tepe,
                                               ln1_g, ln1_b, x1);

  ffn_ln2_kernel<<<ROWBLK, 256, 0, stream>>>(x1, w1p, ff_b1, w2p, ff_b2,
                                             ln2_g, ln2_b, out);
  (void)in_sizes; (void)n_in; (void)out_size; (void)ws_size;
}
